// RelationTagger_6150393168593
// MI455X (gfx1250) — compile-verified, matrix operand order checked by reading, not run on hardware
//
#include <hip/hip_runtime.h>
#include <hip/hip_bf16.h>

// ---------------------------------------------------------------------------
// RelationTagger for MI455X (gfx1250, wave32, WMMA, async-to-LDS staging)
// C=2, F=32, H=1024, B=16, S=2048, SF=S+F=2080
// score[b,c,i,j] = (Q'[c,b,i,:] . key[c,b,j,:]) * vsum[c,b,j]
//   Q' rows 0..S-1   = hidden @ Wq[c]^T
//   Q' rows S..S+F-1 = emb[c]               (batch-broadcast)
//   key              = hidden @ Wk[c]^T
//   vsum[c,b,s]      = hidden[b,s,:] . (sum_o Wv[c,o,:])  <-- value GEMM collapsed
// ---------------------------------------------------------------------------

typedef __attribute__((ext_vector_type(16))) __bf16 v16bf;
typedef __attribute__((ext_vector_type(8)))  float  v8f;

#define C_   2
#define F_   32
#define H_   1024
#define B_   16
#define S_   2048
#define SF_  2080
#define BS_  (B_ * S_)   // 32768
#define KS_  64          // K-step per LDS stage (2 WMMA k-slices)

// round-to-nearest-even f32 -> bf16 (bit pattern)
__device__ __forceinline__ unsigned short f2bf_u(float f) {
    union { float f; unsigned u; } cv; cv.f = f;
    unsigned u = cv.u;
    return (unsigned short)((u + 0x7FFFu + ((u >> 16) & 1u)) >> 16);
}

// gfx1250 async DMA: global -> LDS, 16B per lane, tracked by ASYNCcnt.
// lds_off is the wave-relative LDS byte address (low 32 bits of generic ptr).
__device__ __forceinline__ void async_b128(unsigned lds_off, const void* gptr) {
    asm volatile("global_load_async_to_lds_b128 %0, %1, off"
                 :: "v"(lds_off), "v"((unsigned long long)(size_t)gptr)
                 : "memory");
}
__device__ __forceinline__ void wait_async0() {
    asm volatile("s_wait_asynccnt 0x0" ::: "memory");
}
__device__ __forceinline__ unsigned lds_off32(const void* p) {
    return (unsigned)(size_t)p;   // flat LDS aperture: addr[31:0] == LDS offset
}

// ---------------------------------------------------------------------------
// 1) generic f32 -> bf16 convert (4 elems / thread)
// ---------------------------------------------------------------------------
__global__ __launch_bounds__(256) void f32_to_bf16_kernel(
    const float* __restrict__ src, unsigned short* __restrict__ dst, int n) {
    int base = (blockIdx.x * 256 + threadIdx.x) * 4;
    if (base + 3 < n) {
        float4 v = *reinterpret_cast<const float4*>(src + base);
        dst[base + 0] = f2bf_u(v.x);
        dst[base + 1] = f2bf_u(v.y);
        dst[base + 2] = f2bf_u(v.z);
        dst[base + 3] = f2bf_u(v.w);
    } else {
        for (int j = 0; j < 4; ++j)
            if (base + j < n) dst[base + j] = f2bf_u(src[base + j]);
    }
}

// ---------------------------------------------------------------------------
// 2a) wv_colsum[c,h] = sum_o Wv[c,o,h]
// ---------------------------------------------------------------------------
__global__ __launch_bounds__(256) void wv_colsum_kernel(
    const float* __restrict__ Wv, float* __restrict__ wvcol) {
    int idx = blockIdx.x * 256 + threadIdx.x;   // 0 .. C*H-1
    int c = idx >> 10, h = idx & (H_ - 1);
    const float* base = Wv + (size_t)c * H_ * H_ + h;
    float a = 0.f;
    for (int o = 0; o < H_; ++o) a += base[(size_t)o * H_];
    wvcol[idx] = a;
}

// ---------------------------------------------------------------------------
// 2b) vsum[c, b*S+s] = hidden[b,s,:] . wvcol[c,:]   (f32, wave per row)
// ---------------------------------------------------------------------------
__global__ __launch_bounds__(256) void vsum_kernel(
    const float* __restrict__ hid, const float* __restrict__ wvcol,
    float* __restrict__ vsum) {
    int wave = threadIdx.x >> 5, lane = threadIdx.x & 31;
    int row  = blockIdx.x * 8 + wave;            // 0 .. BS-1
    const float* hrow = hid + (size_t)row * H_;
    float a0 = 0.f, a1 = 0.f;
    for (int t = 0; t < H_ / 32; ++t) {
        float hv = hrow[lane + t * 32];
        a0 += hv * wvcol[lane + t * 32];
        a1 += hv * wvcol[H_ + lane + t * 32];
    }
    for (int off = 16; off > 0; off >>= 1) {     // wave32 reduction
        a0 += __shfl_down(a0, off, 32);
        a1 += __shfl_down(a1, off, 32);
    }
    if (lane == 0) {
        vsum[row]       = a0;
        vsum[BS_ + row] = a1;
    }
}

// ---------------------------------------------------------------------------
// 3) broadcast emb (bf16) into Q' tail rows for every (c,b)
// ---------------------------------------------------------------------------
__global__ __launch_bounds__(256) void emb_broadcast_kernel(
    const float* __restrict__ emb, unsigned short* __restrict__ qfull) {
    int idx = blockIdx.x * 256 + threadIdx.x;    // C*B*F*H
    int h = idx & (H_ - 1);
    int f = (idx >> 10) & (F_ - 1);
    int b = (idx >> 15) & (B_ - 1);
    int c = idx >> 19;
    size_t dst = ((size_t)(c * B_ + b) * SF_ + S_ + f) * H_ + h;
    size_t src = ((size_t)(c * F_ + f)) * H_ + h;
    qfull[dst] = f2bf_u(emb[src]);
}

// ---------------------------------------------------------------------------
// 4) projection GEMM: out[c,m,o] = hiddenB[m,:] . W[c,o,:]   (K=1024, bf16)
//    z in 0..3 : z<2 -> key (Wk), z>=2 -> query into qfull slab (Wq)
//    128x128 block tile, 8 waves (32x64 each), K-step 64,
//    double-buffered LDS fed by global_load_async_to_lds_b128.
// ---------------------------------------------------------------------------
__global__ __launch_bounds__(256) void proj_gemm_kernel(
    const __bf16* __restrict__ hid,      // [BS, H]
    const __bf16* __restrict__ wkB,      // [C, H, H]
    const __bf16* __restrict__ wqB,      // [C, H, H]
    unsigned short* __restrict__ keyB,   // [C, BS, H]
    unsigned short* __restrict__ qfull)  // [C, B, SF, H]
{
    const int z    = blockIdx.z;
    const bool isQ = (z >= C_);
    const int c    = isQ ? z - C_ : z;
    const __bf16* W = (isQ ? wqB : wkB) + (size_t)c * H_ * H_;

    const int nBase = blockIdx.x * 128;
    const int mBase = blockIdx.y * 128;

    __shared__ __align__(32) __bf16 ldsA[2][128 * KS_];   // 2 x 16 KB
    __shared__ __align__(32) __bf16 ldsB[2][128 * KS_];   // 2 x 16 KB

    const int tid   = threadIdx.x;
    const int wave  = tid >> 5, lane = tid & 31;
    const int warpM = (wave & 3) * 32;
    const int warpN = (wave >> 2) * 64;
    const int arow  = lane & 15;
    const int khalf = (lane >> 4) * 16;

    v8f acc[2][4];
    const v8f vzero = {0.f, 0.f, 0.f, 0.f, 0.f, 0.f, 0.f, 0.f};
    for (int mt = 0; mt < 2; ++mt)
        for (int nt = 0; nt < 4; ++nt) acc[mt][nt] = vzero;

    // stage one 128xKS_ tile of A and B into LDS buffer `buf` (async DMA)
    auto stage = [&](int buf, int k0) {
        const __bf16* lA = &ldsA[buf][0];
        const __bf16* lB = &ldsB[buf][0];
        for (int p = 0; p < 4; ++p) {
            int chunk = p * 256 + tid;          // 0..1023 chunks of 8 bf16
            int row   = chunk >> 3;             // 0..127
            int col   = (chunk & 7) * 8;        // bf16 column
            async_b128(lds_off32(lA + row * KS_ + col),
                       hid + (size_t)(mBase + row) * H_ + k0 + col);
            async_b128(lds_off32(lB + row * KS_ + col),
                       W + (size_t)(nBase + row) * H_ + k0 + col);
        }
    };

    stage(0, 0);
    for (int kt = 0; kt < H_ / KS_; ++kt) {
        wait_async0();           // my async batch for buf is in LDS
        __syncthreads();         // everyone's is
        const int buf = kt & 1;
        if (kt + 1 < H_ / KS_) stage(buf ^ 1, (kt + 1) * KS_);   // overlap next tile

        const __bf16* lA = &ldsA[buf][0];
        const __bf16* lB = &ldsB[buf][0];
        for (int ks = 0; ks < KS_ / 32; ++ks) {
            v16bf aF[2], bF[4];
            for (int mt = 0; mt < 2; ++mt)
                aF[mt] = *reinterpret_cast<const v16bf*>(
                    &lA[(warpM + mt * 16 + arow) * KS_ + ks * 32 + khalf]);
            for (int nt = 0; nt < 4; ++nt)
                bF[nt] = *reinterpret_cast<const v16bf*>(
                    &lB[(warpN + nt * 16 + arow) * KS_ + ks * 32 + khalf]);
            for (int mt = 0; mt < 2; ++mt)
                for (int nt = 0; nt < 4; ++nt)
                    acc[mt][nt] = __builtin_amdgcn_wmma_f32_16x16x32_bf16(
                        false, aF[mt], false, bF[nt], (short)0, acc[mt][nt],
                        false, false);
        }
    }

    // epilogue: f32 accum -> bf16 store
    const int mOff = (lane < 16) ? 0 : 8;
    const int nLan = lane & 15;
    for (int mt = 0; mt < 2; ++mt) {
        for (int nt = 0; nt < 4; ++nt) {
            int gn = nBase + warpN + nt * 16 + nLan;
            for (int r = 0; r < 8; ++r) {
                int gm = mBase + warpM + mt * 16 + mOff + r;
                unsigned short v = f2bf_u(acc[mt][nt][r]);
                if (!isQ) {
                    keyB[((size_t)c * BS_ + gm) * H_ + gn] = v;
                } else {
                    int b = gm >> 11, s = gm & (S_ - 1);
                    qfull[((size_t)(c * B_ + b) * SF_ + s) * H_ + gn] = v;
                }
            }
        }
    }
}

// ---------------------------------------------------------------------------
// 5) attention GEMM + fused vsum scale, f32 output
//    per (c,b): D[i,j] = (Q'[i,:] . key[j,:]) * vsum[c,b,j]
//    grid: x = N/128 = 16, y = ceil(SF/128) = 17, z = c*B+b = 32
// ---------------------------------------------------------------------------
__global__ __launch_bounds__(256) void attn_gemm_kernel(
    const __bf16* __restrict__ qfull,    // [C, B, SF, H]
    const __bf16* __restrict__ keyB,     // [C, BS, H]
    const float* __restrict__ vsum,      // [C, BS]
    float* __restrict__ out)             // [B, C, SF, S]
{
    const int cb = blockIdx.z;
    const int c  = cb >> 4, b = cb & (B_ - 1);
    const __bf16* Abase = qfull + (size_t)(c * B_ + b) * SF_ * H_;
    const __bf16* Bbase = keyB + ((size_t)c * BS_ + (size_t)b * S_) * H_;
    const float*  vbase = vsum + (size_t)c * BS_ + (size_t)b * S_;

    const int nBase = blockIdx.x * 128;
    const int mBase = blockIdx.y * 128;

    __shared__ __align__(32) __bf16 ldsA[2][128 * KS_];
    __shared__ __align__(32) __bf16 ldsB[2][128 * KS_];

    const int tid   = threadIdx.x;
    const int wave  = tid >> 5, lane = tid & 31;
    const int warpM = (wave & 3) * 32;
    const int warpN = (wave >> 2) * 64;
    const int arow  = lane & 15;
    const int khalf = (lane >> 4) * 16;

    v8f acc[2][4];
    const v8f vzero = {0.f, 0.f, 0.f, 0.f, 0.f, 0.f, 0.f, 0.f};
    for (int mt = 0; mt < 2; ++mt)
        for (int nt = 0; nt < 4; ++nt) acc[mt][nt] = vzero;

    auto stage = [&](int buf, int k0) {
        const __bf16* lA = &ldsA[buf][0];
        const __bf16* lB = &ldsB[buf][0];
        for (int p = 0; p < 4; ++p) {
            int chunk = p * 256 + tid;
            int row   = chunk >> 3;
            int col   = (chunk & 7) * 8;
            int ga    = mBase + row; if (ga > SF_ - 1) ga = SF_ - 1;  // clamp tail
            async_b128(lds_off32(lA + row * KS_ + col),
                       Abase + (size_t)ga * H_ + k0 + col);
            async_b128(lds_off32(lB + row * KS_ + col),
                       Bbase + (size_t)(nBase + row) * H_ + k0 + col);
        }
    };

    stage(0, 0);
    for (int kt = 0; kt < H_ / KS_; ++kt) {
        wait_async0();
        __syncthreads();
        const int buf = kt & 1;
        if (kt + 1 < H_ / KS_) stage(buf ^ 1, (kt + 1) * KS_);

        const __bf16* lA = &ldsA[buf][0];
        const __bf16* lB = &ldsB[buf][0];
        for (int ks = 0; ks < KS_ / 32; ++ks) {
            v16bf aF[2], bF[4];
            for (int mt = 0; mt < 2; ++mt)
                aF[mt] = *reinterpret_cast<const v16bf*>(
                    &lA[(warpM + mt * 16 + arow) * KS_ + ks * 32 + khalf]);
            for (int nt = 0; nt < 4; ++nt)
                bF[nt] = *reinterpret_cast<const v16bf*>(
                    &lB[(warpN + nt * 16 + arow) * KS_ + ks * 32 + khalf]);
            for (int mt = 0; mt < 2; ++mt)
                for (int nt = 0; nt < 4; ++nt)
                    acc[mt][nt] = __builtin_amdgcn_wmma_f32_16x16x32_bf16(
                        false, aF[mt], false, bF[nt], (short)0, acc[mt][nt],
                        false, false);
        }
    }

    // epilogue: scale by vsum[c,b,j], write f32 score[b,c,i,j]
    const int mOff = (lane < 16) ? 0 : 8;
    const int nLan = lane & 15;
    float vs[4];
    for (int nt = 0; nt < 4; ++nt)
        vs[nt] = vbase[nBase + warpN + nt * 16 + nLan];

    float* obase = out + (size_t)(b * C_ + c) * SF_ * S_;
    for (int mt = 0; mt < 2; ++mt) {
        for (int nt = 0; nt < 4; ++nt) {
            int j = nBase + warpN + nt * 16 + nLan;
            for (int r = 0; r < 8; ++r) {
                int i = mBase + warpM + mt * 16 + mOff + r;
                if (i < SF_)
                    obase[(size_t)i * S_ + j] = acc[mt][nt][r] * vs[nt];
            }
        }
    }
}

// ---------------------------------------------------------------------------
// launch
// ---------------------------------------------------------------------------
extern "C" void kernel_launch(void* const* d_in, const int* in_sizes, int n_in,
                              void* d_out, int out_size, void* d_ws, size_t ws_size,
                              hipStream_t stream) {
    (void)in_sizes; (void)n_in; (void)out_size; (void)ws_size;
    const float* hidden = (const float*)d_in[0];   // [B,S,H]
    const float* emb    = (const float*)d_in[1];   // [C,F,H]
    const float* Wk     = (const float*)d_in[2];   // [C,H,H]
    const float* Wq     = (const float*)d_in[3];   // [C,H,H]
    const float* Wv     = (const float*)d_in[4];   // [C,H,H]
    float* out = (float*)d_out;                    // [B,C,SF,S]

    // workspace carve-up (all sizes are multiples of 256 B)
    char* ws = (char*)d_ws;
    size_t off = 0;
    unsigned short* hiddenB = (unsigned short*)(ws + off); off += (size_t)BS_ * H_ * 2;           // 64 MiB
    unsigned short* wkB     = (unsigned short*)(ws + off); off += (size_t)C_ * H_ * H_ * 2;       // 4 MiB
    unsigned short* wqB     = (unsigned short*)(ws + off); off += (size_t)C_ * H_ * H_ * 2;       // 4 MiB
    float*          wvcol   = (float*)(ws + off);          off += (size_t)C_ * H_ * 4;            // 8 KiB
    float*          vsum    = (float*)(ws + off);          off += (size_t)C_ * BS_ * 4;           // 256 KiB
    unsigned short* keyB    = (unsigned short*)(ws + off); off += (size_t)C_ * BS_ * H_ * 2;      // 128 MiB
    unsigned short* qfull   = (unsigned short*)(ws + off); off += (size_t)C_ * B_ * SF_ * H_ * 2; // 130 MiB

    // 1) converts
    {
        int n = BS_ * H_;
        f32_to_bf16_kernel<<<(n / 4 + 255) / 256, 256, 0, stream>>>(hidden, hiddenB, n);
        n = C_ * H_ * H_;
        f32_to_bf16_kernel<<<(n / 4 + 255) / 256, 256, 0, stream>>>(Wk, wkB, n);
        f32_to_bf16_kernel<<<(n / 4 + 255) / 256, 256, 0, stream>>>(Wq, wqB, n);
    }
    // 2) collapsed value path
    wv_colsum_kernel<<<(C_ * H_) / 256, 256, 0, stream>>>(Wv, wvcol);
    vsum_kernel<<<BS_ / 8, 256, 0, stream>>>(hidden, wvcol, vsum);
    // 3) emb -> Q' tail rows
    emb_broadcast_kernel<<<(C_ * B_ * F_ * H_) / 256, 256, 0, stream>>>(emb, qfull);
    // 4) key + query projections (WMMA, async-LDS pipeline)
    proj_gemm_kernel<<<dim3(H_ / 128, BS_ / 128, 2 * C_), 256, 0, stream>>>(
        (const __bf16*)hiddenB, (const __bf16*)wkB, (const __bf16*)wqB, keyB, qfull);
    // 5) attention + fused vsum scale (WMMA, async-LDS pipeline)
    attn_gemm_kernel<<<dim3(S_ / 128, (SF_ + 127) / 128, C_ * B_), 256, 0, stream>>>(
        (const __bf16*)qfull, (const __bf16*)keyB, vsum, out);
}
